// RefinementLayer1_48945447306023
// MI455X (gfx1250) — compile-verified, matrix operand order checked by reading, not run on hardware
//
#include <hip/hip_runtime.h>

typedef _Float16 v16h __attribute__((ext_vector_type(16)));
typedef _Float16 v8h  __attribute__((ext_vector_type(8)));
typedef float    v8f  __attribute__((ext_vector_type(8)));

#define B_  2
#define L_  192
#define D_  768
#define NC_ 4
#define H_  256
#define KTRUE (H_ + NC_)   // 260 real K for mid layer
#define KP    288          // padded K (multiple of 32)
#define KC    (KP / 32)    // 9 k-chunks (mid layer)

#define NPROJ (H_ + NC_ * H_)   // 1280 projection output columns
#define KC2   (D_ / 32)         // 24 k-chunks (projection)
#define CT2   (NPROJ / 16)      // 80 column tiles (projection)
#define ROWS  (B_ * L_)         // 384 rows

// ---- Workspace layout (bytes) ----
#define BW_HALVES (NC_ * KC * 16 * 32 * 16)        // 294912 (Wmid f16 B-layout)
#define WS_BW_OFF   0
#define WC_HALVES (KC2 * CT2 * 32 * 16)            // 983040 ([Wp|Wa] f16 B-layout)
#define WS_WC_OFF   (BW_HALVES * 2)                // 589824
#define SQ_HALVES (ROWS * D_)                      // 294912 (seq f16)
#define WS_SQ_OFF   (WS_WC_OFF + WC_HALVES * 2)    // 2555904
#define WS_HP_OFF   (WS_SQ_OFF + SQ_HALVES * 2)    // 3145728
#define HP_FLOATS (ROWS * H_)                      // 98304
#define WS_HA_OFF   (WS_HP_OFF + HP_FLOATS * 4)    // 3538944
#define HA_FLOATS (ROWS * NC_ * H_)                // 393216
// total ws used ~= 4.9 MB

// Hardware tanh (TRANS op on CDNA5, co-executes with WMMA).
__device__ __forceinline__ float fast_tanh(float x) {
#if __has_builtin(__builtin_amdgcn_tanh_f32)
  return __builtin_amdgcn_tanh_f32(x);
#else
  float r;
  // TRANS hazard rule: 1 V_NOP before result use (ISA §7.4).
  asm volatile("v_tanh_f32 %0, %1\n\tv_nop" : "=v"(r) : "v"(x));
  return r;
#endif
}

// ---------------------------------------------------------------------------
// Kernel A: repack Wmid [NC, 260, 256] f32 -> padded f16 WMMA-B layout:
// Bws[n][kc][ct][lane][16 halves]; K >= 260 zero-padded.
//   lane < 16 : col = ct*16 + lane,      halves: K = kc*32 + {0..7,16..23}
//   lane >= 16: col = ct*16 + lane - 16, halves: K = kc*32 + {8..15,24..31}
// ---------------------------------------------------------------------------
__global__ void __launch_bounds__(256)
prep_wmid_kernel(const float* __restrict__ Wmid, _Float16* __restrict__ Bws) {
  int idx = blockIdx.x * blockDim.x + threadIdx.x;
  if (idx >= BW_HALVES) return;
  int i    = idx & 15;
  int lane = (idx >> 4) & 31;
  int ct   = (idx >> 9) & 15;
  int rest = idx >> 13;          // n*KC + kc
  int kc   = rest % KC;
  int n    = rest / KC;
  int col  = ct * 16 + (lane & 15);
  int k;
  if (lane < 16) k = kc * 32 + ((i < 8) ? i       : (16 + i - 8));
  else           k = kc * 32 + ((i < 8) ? (8 + i) : (24 + i - 8));
  float v = (k < KTRUE) ? Wmid[((size_t)n * KTRUE + k) * H_ + col] : 0.0f;
  Bws[idx] = (_Float16)v;
}

// ---------------------------------------------------------------------------
// Kernel B: repack [Wp | Wa] (768 x 1280) f32 -> f16 WMMA-B layout:
// Wc[kc][ct][lane][16 halves]   (no K padding needed: 768 % 32 == 0)
// ---------------------------------------------------------------------------
__global__ void __launch_bounds__(256)
prep_wcat_kernel(const float* __restrict__ Wp, const float* __restrict__ Wa,
                 _Float16* __restrict__ Wc) {
  int idx = blockIdx.x * blockDim.x + threadIdx.x;
  if (idx >= WC_HALVES) return;
  int i    = idx & 15;
  int lane = (idx >> 4) & 31;
  int rest = idx >> 9;           // kc*CT2 + ct
  int ct   = rest % CT2;
  int kc   = rest / CT2;
  int col  = ct * 16 + (lane & 15);       // 0..1279
  int k;
  if (lane < 16) k = kc * 32 + ((i < 8) ? i       : (16 + i - 8));
  else           k = kc * 32 + ((i < 8) ? (8 + i) : (24 + i - 8));
  float v = (col < H_) ? Wp[(size_t)k * H_ + col]
                       : Wa[(size_t)k * (NC_ * H_) + (col - H_)];
  Wc[idx] = (_Float16)v;
}

// Kernel C: seq f32 -> f16 (row-major, same indexing)
__global__ void __launch_bounds__(256)
prep_seqh_kernel(const float* __restrict__ seq, _Float16* __restrict__ seqh) {
  int idx = blockIdx.x * blockDim.x + threadIdx.x;
  if (idx < SQ_HALVES) seqh[idx] = (_Float16)seq[idx];
}

// ---------------------------------------------------------------------------
// Kernel D: projection GEMM via WMMA.
// [hp | ha](row, col) = seq(row,:) @ [Wp|Wa](:, col) + [bp|ba](col)
// Grid: (NPROJ/64 = 20, ROWS/16 = 24); block = 128 (4 waves).
// Each wave owns one 16x16 output tile; K = 768 in 24 chunks.
// ---------------------------------------------------------------------------
__global__ void __launch_bounds__(128)
proj_wmma_kernel(const _Float16* __restrict__ seqh,
                 const _Float16* __restrict__ Wc,
                 const float* __restrict__ bp, const float* __restrict__ ba,
                 float* __restrict__ hp, float* __restrict__ ha) {
  const int tid  = threadIdx.x;
  const int wave = tid >> 5;
  const int lane = tid & 31;
  const int ct   = blockIdx.x * 4 + wave;   // column tile (0..79)
  const int row0 = blockIdx.y * 16;

  const int rowm = lane & 15;
  const int koff = (lane >> 4) * 8;

  v8f acc = {0.f, 0.f, 0.f, 0.f, 0.f, 0.f, 0.f, 0.f};
  const _Float16* arow = seqh + (size_t)(row0 + rowm) * D_;

  #pragma unroll
  for (int kc = 0; kc < KC2; ++kc) {
    v8h lo = *(const v8h*)(arow + kc * 32 + koff);
    v8h hi = *(const v8h*)(arow + kc * 32 + koff + 16);
    v16h a = __builtin_shufflevector(lo, hi,
                                     0, 1, 2, 3, 4, 5, 6, 7,
                                     8, 9, 10, 11, 12, 13, 14, 15);
    v16h b = *(const v16h*)(Wc + (((size_t)kc * CT2 + ct) * 32 + lane) * 16);
    acc = __builtin_amdgcn_wmma_f32_16x16x32_f16(
        false, a, false, b, (short)0, acc, false, false);
  }

  const int colN = lane & 15;
  const int half = lane >> 4;
  const int col  = ct * 16 + colN;                 // 0..1279
  const float bias = (col < H_) ? bp[col] : ba[col - H_];
  #pragma unroll
  for (int r = 0; r < 8; ++r) {
    const int row = row0 + r + 8 * half;
    const float v = acc[r] + bias;
    if (col < H_) hp[(size_t)row * H_ + col] = v;
    else          ha[(size_t)row * (NC_ * H_) + (col - H_)] = v;
  }
}

// ---------------------------------------------------------------------------
// Kernel E: fused  tanh(h_p+h_a)‖bs  ->  WMMA x Wmid  ->  tanh(+bmid) . Wout
// Grid: (a_tile=12, p=192, b*NC=8).  Block: 128 threads = 4 waves.
// Each wave owns 4 column tiles (64 h' columns); K loop = 9 chunks of 32.
// ---------------------------------------------------------------------------
__global__ void __launch_bounds__(128)
fused_mid_out_kernel(const float* __restrict__ hp, const float* __restrict__ ha,
                     const float* __restrict__ base_score,
                     const _Float16* __restrict__ Bws,
                     const float* __restrict__ bmid, const float* __restrict__ Wout,
                     float* __restrict__ out) {
  __shared__ alignas(16) _Float16 Alds[16][KP];
  __shared__ float red[4][16];

  const int at = blockIdx.x;       // a-tile index (0..11)
  const int p  = blockIdx.y;       // predicate index (0..191)
  const int bz = blockIdx.z;       // b*NC + n
  const int b  = bz >> 2;
  const int n  = bz & 3;
  const int a0 = at * 16;
  const int tid = threadIdx.x;

  const float* hprow = hp + ((size_t)(b * L_ + p)) * H_;

  // --- Build A tile in LDS: rows = 16 argument positions, cols = padded K ---
  #pragma unroll
  for (int i = 0; i < 32; ++i) {               // 16x256 tanh values
    int e   = i * 128 + tid;
    int row = e >> 8;
    int col = e & 255;
    float v = hprow[col] +
              ha[(((size_t)(b * L_ + (a0 + row)) * NC_) + n) * H_ + col];
    Alds[row][col] = (_Float16)fast_tanh(v);
  }
  if (tid < 64) {                              // base_score cols 256..259
    int row = tid >> 2, j = tid & 3;
    Alds[row][H_ + j] =
        (_Float16)base_score[(((size_t)(b * L_ + p) * NC_) + j) * L_ + (a0 + row)];
  }
  for (int e = tid; e < 16 * (KP - KTRUE); e += 128) {  // zero pad 260..287
    int row = e / (KP - KTRUE);
    int col = KTRUE + e % (KP - KTRUE);
    Alds[row][col] = (_Float16)0.0f;
  }
  __syncthreads();

  const int wave = tid >> 5;
  const int lane = tid & 31;
  const int rowm = lane & 15;          // A-matrix row this lane carries
  const int koff = (lane >> 4) * 8;    // half-wave K offset within chunk

  v8f acc[4];
  const v8f vzero = {0.f, 0.f, 0.f, 0.f, 0.f, 0.f, 0.f, 0.f};
  #pragma unroll
  for (int t = 0; t < 4; ++t) acc[t] = vzero;

  const _Float16* Bbase = Bws + (size_t)n * (KC * 16 * 32 * 16);

  #pragma unroll
  for (int kc = 0; kc < KC; ++kc) {
    v8h lo = *(const v8h*)&Alds[rowm][kc * 32 + koff];
    v8h hi = *(const v8h*)&Alds[rowm][kc * 32 + koff + 16];
    v16h a = __builtin_shufflevector(lo, hi,
                                     0, 1, 2, 3, 4, 5, 6, 7,
                                     8, 9, 10, 11, 12, 13, 14, 15);
    #pragma unroll
    for (int t = 0; t < 4; ++t) {
      const int ct = wave * 4 + t;
      v16h bv = *(const v16h*)(Bbase + (((size_t)kc * 16 + ct) * 32 + lane) * 16);
      acc[t] = __builtin_amdgcn_wmma_f32_16x16x32_f16(
          false, a, false, bv, (short)0, acc[t], false, false);
    }
  }

  // --- Epilogue: out_partial(row) = sum_h' tanh(mid + bmid) * Wout ---
  const int colN = lane & 15;          // h' column within tile
  const int half = lane >> 4;          // 0: rows 0..7, 1: rows 8..15
  float partial[8];
  #pragma unroll
  for (int r = 0; r < 8; ++r) partial[r] = 0.f;

  #pragma unroll
  for (int t = 0; t < 4; ++t) {
    const int hcol = (wave * 4 + t) * 16 + colN;
    const float bm = bmid[n * H_ + hcol];
    const float wo = Wout[n * H_ + hcol];
    #pragma unroll
    for (int r = 0; r < 8; ++r)
      partial[r] += fast_tanh(acc[t][r] + bm) * wo;
  }
  // reduce the 16 columns held across each half-wave
  #pragma unroll
  for (int r = 0; r < 8; ++r) {
    #pragma unroll
    for (int m = 1; m < 16; m <<= 1)
      partial[r] += __shfl_xor(partial[r], m, 32);
  }
  if (colN == 0) {
    #pragma unroll
    for (int r = 0; r < 8; ++r) red[wave][r + 8 * half] = partial[r];
  }
  __syncthreads();

  if (tid < 16) {  // combine 4 waves; write out[b, p, n, a]
    float s = red[0][tid] + red[1][tid] + red[2][tid] + red[3][tid];
    out[(((size_t)(b * L_ + p) * NC_) + n) * L_ + (a0 + tid)] = s;
  }
}

// ---------------------------------------------------------------------------
extern "C" void kernel_launch(void* const* d_in, const int* in_sizes, int n_in,
                              void* d_out, int out_size, void* d_ws, size_t ws_size,
                              hipStream_t stream) {
  const float* seq  = (const float*)d_in[0];
  const float* bsc  = (const float*)d_in[1];
  const float* Wp   = (const float*)d_in[2];
  const float* bp   = (const float*)d_in[3];
  const float* Wa   = (const float*)d_in[4];
  const float* ba   = (const float*)d_in[5];
  const float* Wmid = (const float*)d_in[6];
  const float* bmid = (const float*)d_in[7];
  const float* Wout = (const float*)d_in[8];
  float* out = (float*)d_out;

  char* ws = (char*)d_ws;
  _Float16* Bws  = (_Float16*)(ws + WS_BW_OFF);
  _Float16* Wc   = (_Float16*)(ws + WS_WC_OFF);
  _Float16* seqh = (_Float16*)(ws + WS_SQ_OFF);
  float*    hp   = (float*)(ws + WS_HP_OFF);
  float*    ha   = (float*)(ws + WS_HA_OFF);

  prep_wmid_kernel<<<BW_HALVES / 256, 256, 0, stream>>>(Wmid, Bws);
  prep_wcat_kernel<<<WC_HALVES / 256, 256, 0, stream>>>(Wp, Wa, Wc);
  prep_seqh_kernel<<<SQ_HALVES / 256, 256, 0, stream>>>(seq, seqh);

  dim3 gproj(NPROJ / 64, ROWS / 16);   // 20 x 24 = 480 workgroups
  proj_wmma_kernel<<<gproj, 128, 0, stream>>>(seqh, Wc, bp, ba, hp, ha);

  dim3 grid(L_ / 16, L_, B_ * NC_);    // 12 x 192 x 8 = 18432 workgroups
  fused_mid_out_kernel<<<grid, 128, 0, stream>>>(hp, ha, bsc, Bws, bmid, Wout, out);
}
